// MyGAE_55456617726631
// MI455X (gfx1250) — matrix operand order repaired and checked by reading, not compiled, not run on hardware
//
#include <hip/hip_runtime.h>
#include <hip/hip_bf16.h>
#include <stdint.h>

#define N_NODES 16384
#define N_EDGES 524288
#define F_IN    512
#define H1      256
#define H2      64

typedef __attribute__((ext_vector_type(16))) __bf16        v16bf;
typedef __attribute__((ext_vector_type(8)))  float         v8f;
typedef __attribute__((ext_vector_type(4)))  unsigned int  v4u;

union Frag16 { v16bf v; v4u q[2]; unsigned int u[8]; };

// round-to-nearest-even f32 -> bf16, packed pair into one dword ([15:0]=lo)
__device__ __forceinline__ unsigned int pack2_bf16(float lo, float hi) {
    union { float f; unsigned int u; } a, b;
    a.f = lo; b.f = hi;
    unsigned int ua = (a.u + 0x7FFFu + ((a.u >> 16) & 1u)) >> 16;
    unsigned int ub = (b.u + 0x7FFFu + ((b.u >> 16) & 1u)) >> 16;
    return (ua & 0xFFFFu) | (ub << 16);
}

// ---------------- degree / norm ----------------
__global__ void k_init_deg(float* __restrict__ deg) {
    int i = blockIdx.x * blockDim.x + threadIdx.x;
    if (i < N_NODES) deg[i] = 1.0f;   // self-loop weight
}

__global__ void k_edge_deg(float* __restrict__ deg, const int* __restrict__ dst,
                           const float* __restrict__ ew) {
    int e = blockIdx.x * blockDim.x + threadIdx.x;
    if (e < N_EDGES) atomicAdd(&deg[dst[e]], ew[e]);
}

__global__ void k_dinv(const float* __restrict__ deg, float* __restrict__ dinv) {
    int i = blockIdx.x * blockDim.x + threadIdx.x;
    if (i < N_NODES) {
        float d = deg[i];
        dinv[i] = (d > 0.0f) ? __frsqrt_rn(d) : 0.0f;
    }
}

// ---------------- f32 -> packed bf16 casts (one-shot, operands stay in L2) ------
__global__ void k_cast_bf16(const float* __restrict__ in, unsigned int* __restrict__ out,
                            int n2) {
    int t = blockIdx.x * blockDim.x + threadIdx.x;
    if (t < n2) {
        const float2 f = *(const float2*)(in + 2 * (size_t)t);
        out[t] = pack2_bf16(f.x, f.y);
    }
}

__global__ void k_relu_cast_bf16(const float* __restrict__ in,
                                 unsigned int* __restrict__ out, int n2) {
    int t = blockIdx.x * blockDim.x + threadIdx.x;
    if (t < n2) {
        const float2 f = *(const float2*)(in + 2 * (size_t)t);
        out[t] = pack2_bf16(fmaxf(f.x, 0.0f), fmaxf(f.y, 0.0f));
    }
}

// W: [R,C] f32 row-major  ->  Wt: [C, R/2] dwords, bf16 pairs along R (K-major)
__global__ void k_transpose_cast_bf16(const float* __restrict__ W,
                                      unsigned int* __restrict__ Wt, int R, int C) {
    int t = blockIdx.x * blockDim.x + threadIdx.x;
    int r2d = R >> 1;
    if (t < C * r2d) {
        int c = t / r2d, r2 = t % r2d;
        Wt[t] = pack2_bf16(W[(size_t)(2 * r2) * C + c], W[(size_t)(2 * r2 + 1) * C + c]);
    }
}

// ---------------- WMMA GEMM: C[M,N] = A[M,K] @ B[N,K]^T  (packed-bf16 operands) --
// Both operands K-major: every fragment = two contiguous b128 loads, zero VALU
// conversion in the loop. One wave computes a (16*MT) x (16*NT) tile; each B
// fragment feeds MT WMMAs (1.5 b128 loads per WMMA at MT=2, NT=4).
template <int K, int MT, int NT>
__global__ void k_gemm_nt_bf16(const unsigned int* __restrict__ A,   // [M, K/2]
                               const unsigned int* __restrict__ B,   // [N, K/2]
                               float* __restrict__ C, int M, int N) {
    constexpr int KD = K / 2;                 // dwords per row
    const int lane = threadIdx.x & 31;
    const int wave = threadIdx.x >> 5;
    const int stripsN = N / (16 * NT);
    const int strip = blockIdx.x * 8 + wave;
    if (strip >= (M / (16 * MT)) * stripsN) return;  // uniform across wave
    const int sm = strip / stripsN, sn = strip % stripsN;
    const int hf = lane >> 4, l15 = lane & 15;

    const unsigned int* arow[MT];
#pragma unroll
    for (int mt = 0; mt < MT; ++mt)
        arow[mt] = A + (size_t)(((sm * MT + mt) << 4) + l15) * KD;
    const unsigned int* brow[NT];
#pragma unroll
    for (int nt = 0; nt < NT; ++nt)
        brow[nt] = B + (size_t)(((sn * NT + nt) << 4) + l15) * KD;

    v8f acc[MT][NT] = {};

#pragma unroll
    for (int k0 = 0; k0 < KD; k0 += 16) {     // 32 K-elements per chunk
        Frag16 fa[MT];
#pragma unroll
        for (int mt = 0; mt < MT; ++mt) {
            fa[mt].q[0] = *(const v4u*)(arow[mt] + k0 + 4 * hf);     // v=0..3
            fa[mt].q[1] = *(const v4u*)(arow[mt] + k0 + 8 + 4 * hf); // v=4..7
        }
#pragma unroll
        for (int nt = 0; nt < NT; ++nt) {
            Frag16 fb;
            fb.q[0] = *(const v4u*)(brow[nt] + k0 + 4 * hf);
            fb.q[1] = *(const v4u*)(brow[nt] + k0 + 8 + 4 * hf);
#pragma unroll
            for (int mt = 0; mt < MT; ++mt)
                acc[mt][nt] = __builtin_amdgcn_wmma_f32_16x16x32_bf16(
                    false, fa[mt].v, false, fb.v, (short)0, acc[mt][nt], false, false);
        }
    }
    // C layout: VGPR v -> row = v + 8*half, col = lane&15
#pragma unroll
    for (int mt = 0; mt < MT; ++mt) {
#pragma unroll
        for (int nt = 0; nt < NT; ++nt) {
            float* crow = C + (size_t)(((sm * MT + mt) << 4) + (hf << 3)) * N
                            + ((sn * NT + nt) << 4) + l15;
#pragma unroll
            for (int v = 0; v < 8; ++v)
                crow[(size_t)v * N] = acc[mt][nt][v];
        }
    }
}

// ---------------- message init: out[i][f] = b[f] + h[i][f]*dinv[i]^2 -------------
__global__ void k_msg_init(const float* __restrict__ h, const float* __restrict__ bias,
                           const float* __restrict__ dinv, float* __restrict__ out, int F) {
    int t = blockIdx.x * blockDim.x + threadIdx.x;
    int i = t / F, f = t % F;
    if (i < N_NODES) {
        float di = dinv[i];
        out[t] = bias[f] + h[t] * di * di;
    }
}

// ---------------- edge scatter: out[dst] += h[src] * (dinv[s]*ew*dinv[d]) --------
__global__ void k_scatter(const float* __restrict__ h, float* __restrict__ out,
                          const int* __restrict__ src, const int* __restrict__ dst,
                          const float* __restrict__ ew, const float* __restrict__ dinv,
                          int F4, int F) {
    long long t = (long long)blockIdx.x * blockDim.x + threadIdx.x;
    int e = (int)(t / F4);
    int c = (int)(t % F4) << 2;
    if (e >= N_EDGES) return;
    int s = src[e], d = dst[e];
    float coef = dinv[s] * ew[e] * dinv[d];
    const float4 hv = *(const float4*)(h + (size_t)s * F + c);
    float* o = out + (size_t)d * F + c;
    atomicAdd(o + 0, hv.x * coef);
    atomicAdd(o + 1, hv.y * coef);
    atomicAdd(o + 2, hv.z * coef);
    atomicAdd(o + 3, hv.w * coef);
}

extern "C" void kernel_launch(void* const* d_in, const int* in_sizes, int n_in,
                              void* d_out, int out_size, void* d_ws, size_t ws_size,
                              hipStream_t stream) {
    const float* x  = (const float*)d_in[0];
    const int*   ei = (const int*)d_in[1];
    const float* ew = (const float*)d_in[2];
    const float* W1 = (const float*)d_in[3];
    const float* b1 = (const float*)d_in[4];
    const float* W2 = (const float*)d_in[5];
    const float* b2 = (const float*)d_in[6];
    const int* src = ei;
    const int* dst = ei + N_EDGES;

    float* adj = (float*)d_out;                          // [N, N]
    float* z   = adj + (size_t)N_NODES * N_NODES;        // [N, H2] (2nd tuple output)

    // workspace layout (all 4-byte units, 16B-aligned blocks)
    float*        ws   = (float*)d_ws;
    float*        deg  = ws;                                        // [N]
    float*        dinv = deg + N_NODES;                             // [N]
    unsigned int* xbf  = (unsigned int*)(dinv + N_NODES);           // [N, F_IN/2]
    unsigned int* w1t  = xbf + (size_t)N_NODES * (F_IN / 2);        // [H1, F_IN/2]
    float*        h    = (float*)(w1t + (size_t)H1 * (F_IN / 2));   // [N, H1]
    float*        a1   = h + (size_t)N_NODES * H1;                  // [N, H1]
    unsigned int* a1bf = (unsigned int*)(a1 + (size_t)N_NODES * H1);// [N, H1/2]
    unsigned int* w2t  = a1bf + (size_t)N_NODES * (H1 / 2);         // [H2, H1/2]
    float*        h2   = (float*)(w2t + (size_t)H2 * (H1 / 2));     // [N, H2]
    unsigned int* zbf  = (unsigned int*)(h2 + (size_t)N_NODES * H2);// [N, H2/2]

    // ---- normalization ----
    k_init_deg<<<N_NODES / 256, 256, 0, stream>>>(deg);
    k_edge_deg<<<N_EDGES / 256, 256, 0, stream>>>(deg, dst, ew);
    k_dinv<<<N_NODES / 256, 256, 0, stream>>>(deg, dinv);

    // ---- operand prep (one-shot casts/transposes) ----
    k_cast_bf16<<<(N_NODES * F_IN / 2) / 256, 256, 0, stream>>>(x, xbf, N_NODES * F_IN / 2);
    k_transpose_cast_bf16<<<(H1 * F_IN / 2) / 256, 256, 0, stream>>>(W1, w1t, F_IN, H1);
    k_transpose_cast_bf16<<<(H2 * H1 / 2) / 256, 256, 0, stream>>>(W2, w2t, H1, H2);

    // ---- layer 1: h = x @ W1 ; a1 = scatter(norm*h) + b1 ; relu fused into cast --
    k_gemm_nt_bf16<F_IN, 2, 4><<<(N_NODES / 32) * (H1 / 64) / 8, 256, 0, stream>>>(
        xbf, w1t, h, N_NODES, H1);
    k_msg_init<<<N_NODES * H1 / 256, 256, 0, stream>>>(h, b1, dinv, a1, H1);
    k_scatter<<<(N_EDGES * (H1 / 4)) / 256, 256, 0, stream>>>(h, a1, src, dst,
                                                              ew, dinv, H1 / 4, H1);
    k_relu_cast_bf16<<<(N_NODES * H1 / 2) / 256, 256, 0, stream>>>(a1, a1bf,
                                                                   N_NODES * H1 / 2);

    // ---- layer 2: h2 = relu(a1) @ W2 ; z = scatter(norm*h2) + b2 ----
    k_gemm_nt_bf16<H1, 2, 4><<<(N_NODES / 32) * (H2 / 64) / 8, 256, 0, stream>>>(
        a1bf, w2t, h2, N_NODES, H2);
    k_msg_init<<<N_NODES * H2 / 256, 256, 0, stream>>>(h2, b2, dinv, z, H2);
    k_scatter<<<(N_EDGES * (H2 / 4)) / 256, 256, 0, stream>>>(h2, z, src, dst,
                                                              ew, dinv, H2 / 4, H2);

    // ---- decoder: adj = z @ z^T  (compute-bound; bf16 z is 2 MB -> L2 resident) --
    k_cast_bf16<<<(N_NODES * H2 / 2) / 256, 256, 0, stream>>>(z, zbf, N_NODES * H2 / 2);
    k_gemm_nt_bf16<H2, 2, 4><<<(N_NODES / 32) * (N_NODES / 64) / 8, 256, 0, stream>>>(
        zbf, zbf, adj, N_NODES, N_NODES);
}
